// MomentInteraction_73821897883896
// MI455X (gfx1250) — matrix operand order repaired and checked by reading, not compiled, not run on hardware
//
#include <hip/hip_runtime.h>

// ---------------------------------------------------------------------------
// Types for CDNA5 WMMA (wave32): A/B = 16 bf16 per lane, C/D = 8 f32 per lane
// ---------------------------------------------------------------------------
typedef __attribute__((ext_vector_type(16))) __bf16 v16bf;
typedef __attribute__((ext_vector_type(8)))  float  v8f;

#define WMMA_BF16(A, B, C) \
  __builtin_amdgcn_wmma_f32_16x16x32_bf16(false, (A), false, (B), (short)0, (C), false, false)

__device__ __forceinline__ float silu_f(float x) {
  return x / (1.0f + __expf(-x));
}

// Non-returning fp32 atomic add, relaxed/agent scope -> global_atomic_add_f32
__device__ __forceinline__ void atomic_add_f32(float* p, float v) {
  __hip_atomic_fetch_add(p, v, __ATOMIC_RELAXED, __HIP_MEMORY_SCOPE_AGENT);
}

// B fragment load: packed layout [nt][kt][lane][16 bf16] -> contiguous 32B per lane
__device__ __forceinline__ v16bf load_bfrag(const __bf16* __restrict__ Wp,
                                            int nt, int KT, int kt, int lane) {
  return *(const v16bf*)(Wp + (((size_t)(nt * KT + kt) << 9) + lane * 16));
}

// ---------------------------------------------------------------------------
// Pack fp32 weight W[K][N] (row-major) into bf16 B-fragment layout.
// B 32x16: lanes 0-15 hold K=kt*32+{0..7,16..23}, lanes 16-31 hold
// K=kt*32+{8..15,24..31}; column N = nt*16 + (lane&15).
// ---------------------------------------------------------------------------
__global__ void pack_weight_bf16(const float* __restrict__ W,
                                 __bf16* __restrict__ dst, int K, int N) {
  int idx = blockIdx.x * blockDim.x + threadIdx.x;
  if (idx >= K * N) return;
  int e    = idx & 15;
  int l    = (idx >> 4) & 31;
  int frag = idx >> 9;
  int KT = K >> 5;
  int nt = frag / KT;
  int kt = frag - nt * KT;
  int n = (nt << 4) + (l & 15);
  int k = (kt << 5) + ((l >> 4) << 3) + (e & 7) + ((e >> 3) << 4);
  dst[idx] = (__bf16)W[k * N + n];
}

__global__ void zero_f4(float* __restrict__ p, int n4) {
  int i = blockIdx.x * blockDim.x + threadIdx.x;
  if (i < n4) ((float4*)p)[i] = make_float4(0.f, 0.f, 0.f, 0.f);
}

// ---------------------------------------------------------------------------
// Edge MLP: h = silu(silu(f@W1+b1)@W2+b2); split into f01|f02.
// One 16-row tile per 256-thread block (8 waves, 2 N-tiles each over N=256).
// ---------------------------------------------------------------------------
__global__ __launch_bounds__(256) void edge_mlp_kernel(
    const float* __restrict__ f,
    const __bf16* __restrict__ W1p, const float* __restrict__ b1,
    const __bf16* __restrict__ W2p, const float* __restrict__ b2,
    float* __restrict__ f01, float* __restrict__ f02) {
  __shared__ float  Ain[16][256];   // fp32 layer-1 input tile (16 KB)
  __shared__ __bf16 Amid[16][256];  // bf16 layer-1 activations (8 KB)

  const int m0  = blockIdx.x * 16;
  const int tid = threadIdx.x;

  // Cooperative load of 16x256 fp32 tile (1024 float4)
  const float4* src4 = (const float4*)(f + (size_t)m0 * 256);
  float4* dst4 = (float4*)&Ain[0][0];
  for (int i = tid; i < 1024; i += 256) dst4[i] = src4[i];
  __syncthreads();

  const int wave = tid >> 5;
  const int lane = tid & 31;
  const int m    = lane & 15;
  const int kh   = (lane < 16) ? 0 : 8;  // A-layout K sub-offset
  const int rh   = (lane < 16) ? 0 : 8;  // C-layout row sub-offset

  // Layer 1: build 8 A fragments (K=256), fp32 -> bf16
  v16bf A1[8];
#pragma unroll
  for (int kt = 0; kt < 8; ++kt) {
    int kb = kt * 32 + kh;
    v16bf a;
#pragma unroll
    for (int e = 0; e < 8; ++e) {
      a[e]     = (__bf16)Ain[m][kb + e];
      a[e + 8] = (__bf16)Ain[m][kb + 16 + e];
    }
    A1[kt] = a;
  }
#pragma unroll
  for (int j = 0; j < 2; ++j) {
    int nt = wave * 2 + j;
    v8f acc = {};
#pragma unroll
    for (int kt = 0; kt < 8; ++kt)
      acc = WMMA_BF16(A1[kt], load_bfrag(W1p, nt, 8, kt, lane), acc);
    int n = nt * 16 + m;
    float bias = b1[n];
#pragma unroll
    for (int v = 0; v < 8; ++v)
      Amid[v + rh][n] = (__bf16)silu_f(acc[v] + bias);
  }
  __syncthreads();

  // Layer 2: A fragments directly from bf16 LDS
  v16bf A2[8];
#pragma unroll
  for (int kt = 0; kt < 8; ++kt) {
    int kb = kt * 32 + kh;
    v16bf a;
#pragma unroll
    for (int e = 0; e < 8; ++e) {
      a[e]     = Amid[m][kb + e];
      a[e + 8] = Amid[m][kb + 16 + e];
    }
    A2[kt] = a;
  }
#pragma unroll
  for (int j = 0; j < 2; ++j) {
    int nt = wave * 2 + j;
    v8f acc = {};
#pragma unroll
    for (int kt = 0; kt < 8; ++kt)
      acc = WMMA_BF16(A2[kt], load_bfrag(W2p, nt, 8, kt, lane), acc);
    int n = nt * 16 + m;
    float bias = b2[n];
    float* dst = (n < 128) ? f01 : f02;
    int nc = n & 127;
#pragma unroll
    for (int v = 0; v < 8; ++v) {
      int row = m0 + v + rh;
      dst[(size_t)row * 128 + nc] = silu_f(acc[v] + bias);
    }
  }
}

// ---------------------------------------------------------------------------
// Triplet kernel: per 16-triplet tile (1 wave each; 4 waves / 128-thread block)
//   prodXX = fXX[ki]*MXX*fXX[ji]  (coalesced into LDS, fp32)
//   sXX = silu(prodXX @ WsXXp + bsXX)   -> bf16 LDS (concat buffer)
//   s = silu(concat @ Wsp + bs)         -> atomic scatter-add into accum[E,128]
// ---------------------------------------------------------------------------
__global__ __launch_bounds__(128) void triplet_kernel(
    const float* __restrict__ f01, const float* __restrict__ f02,
    const float* __restrict__ M01, const float* __restrict__ M02,
    const __bf16* __restrict__ Ws01p, const float* __restrict__ bs01,
    const __bf16* __restrict__ Ws02p, const float* __restrict__ bs02,
    const __bf16* __restrict__ Wsp,   const float* __restrict__ bs,
    const int* __restrict__ idx_ji,   const int* __restrict__ idx_ki,
    float* __restrict__ accum) {
  __shared__ float  prod[4][16][128];  // 32 KB
  __shared__ __bf16 sA[4][16][256];    // 32 KB

  const int wave = threadIdx.x >> 5;
  const int lane = threadIdx.x & 31;
  const int t0   = (blockIdx.x * 4 + wave) * 16;

  float  (*P)[128]  = prod[wave];
  __bf16 (*SA)[256] = sA[wave];

  const int m  = lane & 15;
  const int kh = (lane < 16) ? 0 : 8;
  const int rh = (lane < 16) ? 0 : 8;

  for (int half = 0; half < 2; ++half) {
    const float*  fsrc = half ? f02 : f01;
    const float*  Msrc = half ? M02 : M01;
    const __bf16* Wp   = half ? Ws02p : Ws01p;
    const float*  bsrc = half ? bs02 : bs01;

    __syncthreads();  // protect reuse of prod[] across halves
    // Build product tile: each lane one float4 column-chunk per row
    for (int r = 0; r < 16; ++r) {
      int t  = t0 + r;
      int ji = idx_ji[t];
      int ki = idx_ki[t];
      const float4 a = *(const float4*)(fsrc + (size_t)ki * 128 + lane * 4);
      const float4 g = *(const float4*)(Msrc + (size_t)t  * 128 + lane * 4);
      const float4 b = *(const float4*)(fsrc + (size_t)ji * 128 + lane * 4);
      float4 pr;
      pr.x = a.x * g.x * b.x;
      pr.y = a.y * g.y * b.y;
      pr.z = a.z * g.z * b.z;
      pr.w = a.w * g.w * b.w;
      *(float4*)&P[r][lane * 4] = pr;
    }
    __syncthreads();

    // A fragments (K=128 -> 4 k-tiles), fp32 -> bf16
    v16bf A[4];
#pragma unroll
    for (int kt = 0; kt < 4; ++kt) {
      int kb = kt * 32 + kh;
      v16bf a;
#pragma unroll
      for (int e = 0; e < 8; ++e) {
        a[e]     = (__bf16)P[m][kb + e];
        a[e + 8] = (__bf16)P[m][kb + 16 + e];
      }
      A[kt] = a;
    }
#pragma unroll
    for (int nt = 0; nt < 8; ++nt) {
      v8f acc = {};
#pragma unroll
      for (int kt = 0; kt < 4; ++kt)
        acc = WMMA_BF16(A[kt], load_bfrag(Wp, nt, 4, kt, lane), acc);
      int n = nt * 16 + m;
      float bias = bsrc[n];
#pragma unroll
      for (int v = 0; v < 8; ++v)
        SA[v + rh][half * 128 + n] = (__bf16)silu_f(acc[v] + bias);
    }
  }
  __syncthreads();

  // s = silu(concat(s01,s02) @ Ws + bs), K=256 -> 8 k-tiles, N=128 -> 8 n-tiles
  v16bf AS[8];
#pragma unroll
  for (int kt = 0; kt < 8; ++kt) {
    int kb = kt * 32 + kh;
    v16bf a;
#pragma unroll
    for (int e = 0; e < 8; ++e) {
      a[e]     = SA[m][kb + e];
      a[e + 8] = SA[m][kb + 16 + e];
    }
    AS[kt] = a;
  }
  int ej[8];
#pragma unroll
  for (int v = 0; v < 8; ++v) ej[v] = idx_ji[t0 + v + rh];

#pragma unroll
  for (int nt = 0; nt < 8; ++nt) {
    v8f acc = {};
#pragma unroll
    for (int kt = 0; kt < 8; ++kt)
      acc = WMMA_BF16(AS[kt], load_bfrag(Wsp, nt, 8, kt, lane), acc);
    int n = nt * 16 + m;
    float bias = bs[n];
#pragma unroll
    for (int v = 0; v < 8; ++v) {
      float x = silu_f(acc[v] + bias);
      atomic_add_f32(accum + (size_t)ej[v] * 128 + n, x);  // segment_sum scatter
    }
  }
}

// ---------------------------------------------------------------------------
// Final projection in-place: out = out @ Wout + bout  ([E,128]x[128,128])
// Block owns 16 exclusive rows; stage into LDS, then overwrite.
// ---------------------------------------------------------------------------
__global__ __launch_bounds__(256) void out_gemm_kernel(
    float* __restrict__ out, const __bf16* __restrict__ Woutp,
    const float* __restrict__ bout) {
  __shared__ float Ain[16][128];  // 8 KB

  const int m0  = blockIdx.x * 16;
  const int tid = threadIdx.x;

  const float4* src4 = (const float4*)(out + (size_t)m0 * 128);
  float4* dst4 = (float4*)&Ain[0][0];
  for (int i = tid; i < 512; i += 256) dst4[i] = src4[i];
  __syncthreads();

  const int wave = tid >> 5;
  const int lane = tid & 31;
  const int m  = lane & 15;
  const int kh = (lane < 16) ? 0 : 8;
  const int rh = (lane < 16) ? 0 : 8;

  v16bf A[4];
#pragma unroll
  for (int kt = 0; kt < 4; ++kt) {
    int kb = kt * 32 + kh;
    v16bf a;
#pragma unroll
    for (int e = 0; e < 8; ++e) {
      a[e]     = (__bf16)Ain[m][kb + e];
      a[e + 8] = (__bf16)Ain[m][kb + 16 + e];
    }
    A[kt] = a;
  }
  const int nt = wave;  // 8 waves cover N=128
  v8f acc = {};
#pragma unroll
  for (int kt = 0; kt < 4; ++kt)
    acc = WMMA_BF16(A[kt], load_bfrag(Woutp, nt, 4, kt, lane), acc);
  int n = nt * 16 + m;
  float bias = bout[n];
#pragma unroll
  for (int v = 0; v < 8; ++v)
    out[(size_t)(m0 + v + rh) * 128 + n] = acc[v] + bias;
}

// ---------------------------------------------------------------------------
// Host launcher
// ---------------------------------------------------------------------------
extern "C" void kernel_launch(void* const* d_in, const int* in_sizes, int n_in,
                              void* d_out, int out_size, void* d_ws, size_t ws_size,
                              hipStream_t stream) {
  const float* f    = (const float*)d_in[0];
  const float* M01  = (const float*)d_in[1];
  const float* M02  = (const float*)d_in[2];
  const float* W1   = (const float*)d_in[3];
  const float* b1   = (const float*)d_in[4];
  const float* W2   = (const float*)d_in[5];
  const float* b2   = (const float*)d_in[6];
  const float* Ws01 = (const float*)d_in[7];
  const float* bs01 = (const float*)d_in[8];
  const float* Ws02 = (const float*)d_in[9];
  const float* bs02 = (const float*)d_in[10];
  const float* Ws   = (const float*)d_in[11];
  const float* bs   = (const float*)d_in[12];
  const float* Wout = (const float*)d_in[13];
  const float* bout = (const float*)d_in[14];
  const int* idx_ji = (const int*)d_in[16];
  const int* idx_ki = (const int*)d_in[17];

  const int E = in_sizes[0] / 256;  // 100000
  const int T = in_sizes[1] / 128;  // 1000000

  // Workspace layout
  char* w = (char*)d_ws;
  float* f01 = (float*)w;  w += (size_t)E * 128 * 4;
  float* f02 = (float*)w;  w += (size_t)E * 128 * 4;
  __bf16* W1p   = (__bf16*)w;  w += 256 * 256 * 2;
  __bf16* W2p   = (__bf16*)w;  w += 256 * 256 * 2;
  __bf16* Ws01p = (__bf16*)w;  w += 128 * 128 * 2;
  __bf16* Ws02p = (__bf16*)w;  w += 128 * 128 * 2;
  __bf16* Wsp   = (__bf16*)w;  w += 256 * 128 * 2;
  __bf16* Woutp = (__bf16*)w;  w += 128 * 128 * 2;

  // 1) Pack weights to bf16 WMMA fragment layout
  pack_weight_bf16<<<(256 * 256 + 255) / 256, 256, 0, stream>>>(W1, W1p, 256, 256);
  pack_weight_bf16<<<(256 * 256 + 255) / 256, 256, 0, stream>>>(W2, W2p, 256, 256);
  pack_weight_bf16<<<(128 * 128 + 255) / 256, 256, 0, stream>>>(Ws01, Ws01p, 128, 128);
  pack_weight_bf16<<<(128 * 128 + 255) / 256, 256, 0, stream>>>(Ws02, Ws02p, 128, 128);
  pack_weight_bf16<<<(256 * 128 + 255) / 256, 256, 0, stream>>>(Ws, Wsp, 256, 128);
  pack_weight_bf16<<<(128 * 128 + 255) / 256, 256, 0, stream>>>(Wout, Woutp, 128, 128);

  // 2) Edge MLP (E divisible by 16)
  edge_mlp_kernel<<<E / 16, 256, 0, stream>>>(f, W1p, b1, W2p, b2, f01, f02);

  // 3) Zero output accumulator (used for segment_sum)
  int n4 = E * 128 / 4;
  zero_f4<<<(n4 + 255) / 256, 256, 0, stream>>>((float*)d_out, n4);

  // 4) Triplet interaction + scatter-add (T divisible by 64)
  triplet_kernel<<<T / 64, 128, 0, stream>>>(f01, f02, M01, M02,
                                             Ws01p, bs01, Ws02p, bs02,
                                             Wsp, bs, idx_ji, idx_ki,
                                             (float*)d_out);

  // 5) Final projection in-place on d_out
  out_gemm_kernel<<<E / 16, 256, 0, stream>>>((float*)d_out, Woutp, bout);
}
